// WindowAttention_1769526525987
// MI455X (gfx1250) — compile-verified, matrix operand order checked by reading, not run on hardware
//
#include <hip/hip_runtime.h>
#include <math.h>

// ---------------- types ----------------
typedef __bf16 bf16_t;
typedef __attribute__((ext_vector_type(8)))  __bf16 v8bf;
typedef __attribute__((ext_vector_type(16))) __bf16 v16bf;
typedef __attribute__((ext_vector_type(8)))  float  v8f;

#define B_TOT 1024
#define NTOK  64
#define CCH   512
#define HEADS 16
#define HDIM  32
#define NWIN  64

#define SCALE_F 0.17677669529663687f   // 1/sqrt(32)
#define GAIN_F  0.044194173824159216f  // 1/sqrt(512)

// LDS leading dims (elements). Row strides kept 16B-aligned.
#define LDX  520   // x / O tiles: 64 x 520 bf16
#define LDQ  40    // q/k tiles:   64 x 40 bf16
#define LDVT 72    // v^T tile:    32 x 72 bf16
#define LDP  72    // prob tile:   64 x 72 bf16
#define LDA  68    // score tile:  64 x 68 f32

// float -> bf16, round-to-nearest-even (bit-exact, no reliance on __bf16 arith)
__device__ __forceinline__ bf16_t f2bf(float f) {
    unsigned u = __builtin_bit_cast(unsigned, f);
    u = (u + 0x7FFFu + ((u >> 16) & 1u)) >> 16;
    unsigned short s = (unsigned short)u;
    return __builtin_bit_cast(bf16_t, s);
}

// Build a 16-bit-element WMMA A/B fragment per the CDNA5 16x32 (Mx K) layout:
//   lane<16 : row = row0+lane,    K = k0+0..7  (v0..3) and k0+16..23 (v4..7)
//   lane>=16: row = row0+lane-16, K = k0+8..15 (v0..3) and k0+24..31 (v4..7)
// Two 16-byte contiguous loads per lane (ds_load_b128 from LDS, b128 from global).
__device__ __forceinline__ v16bf load_frag(const bf16_t* base, int ld,
                                           int row0, int k0, int lane) {
    const int r  = row0 + (lane & 15);
    const int kb = k0 + ((lane >> 4) << 3);
    const bf16_t* p = base + r * ld + kb;
    v8bf lo = *(const v8bf*)(p);
    v8bf hi = *(const v8bf*)(p + 16);
    return __builtin_shufflevector(lo, hi,
        0,1,2,3,4,5,6,7,8,9,10,11,12,13,14,15);
}

__device__ __forceinline__ v8f wmma_bf16(v16bf a, v16bf b, v8f c) {
    return __builtin_amdgcn_wmma_f32_16x16x32_bf16(
        /*neg_a=*/false, a, /*neg_b=*/false, b,
        /*c_mod=*/(short)0, c, /*reuse_a=*/false, /*reuse_b=*/false);
}

__device__ __forceinline__ v8f v8f_zero() {
    v8f z = {0.f,0.f,0.f,0.f,0.f,0.f,0.f,0.f};
    return z;
}

// ---------------- kernel 1: weight convert (w * GAIN -> bf16) ----------------
__global__ __launch_bounds__(256)
void cvt_weights_kernel(const float* __restrict__ qw, const float* __restrict__ kw,
                        const float* __restrict__ vw, const float* __restrict__ pw,
                        bf16_t* __restrict__ wsb) {
    int i = blockIdx.x * blockDim.x + threadIdx.x;   // 0 .. 262143
    if (i >= CCH * CCH) return;
    wsb[i]              = f2bf(qw[i] * GAIN_F);
    wsb[CCH*CCH   + i]  = f2bf(kw[i] * GAIN_F);
    wsb[2*CCH*CCH + i]  = f2bf(vw[i] * GAIN_F);
    wsb[3*CCH*CCH + i]  = f2bf(pw[i] * GAIN_F);
}

// ---------------- kernel 2: fused window attention ----------------
__global__ __launch_bounds__(256, 1)
void win_attn_kernel(const float* __restrict__ x,
                     const float* __restrict__ mwin,   // [B,64,1]
                     const float* __restrict__ mask,   // [64,64,64]
                     const bf16_t* __restrict__ wq, const float* __restrict__ qb,
                     const bf16_t* __restrict__ wk, const float* __restrict__ kb,
                     const bf16_t* __restrict__ wv, const float* __restrict__ vb,
                     const bf16_t* __restrict__ wp, const float* __restrict__ pb,
                     float* __restrict__ out,          // [B,64,512]
                     float* __restrict__ mask_new) {   // [B,64]
    extern __shared__ char smem[];
    bf16_t* sX    = (bf16_t*)smem;            // 64 x LDX   (x, bf16)
    bf16_t* sO    = sX  + NTOK * LDX;         // 64 x LDX   (attn output, bf16)
    bf16_t* sQ    = sO  + NTOK * LDX;         // 64 x LDQ
    bf16_t* sK    = sQ  + NTOK * LDQ;         // 64 x LDQ
    bf16_t* sVt   = sK  + NTOK * LDQ;         // 32 x LDVT  (V transposed: [d][tok])
    bf16_t* sP    = sVt + HDIM * LDVT;        // 64 x LDP   (softmax probs, bf16)
    float*  sAttn = (float*)(sP + NTOK * LDP);// 64 x LDA
    float*  sNrm  = sAttn + NTOK * LDA;       // 64
    float*  sBias = sNrm + NTOK;              // 64
    float*  sMn   = sBias + NTOK;             // 1

    const int b    = blockIdx.x;
    const int tid  = threadIdx.x;
    const int lane = tid & 31;
    const int wave = tid >> 5;
    const int l15  = lane & 15;
    const int lhi  = lane >> 4;

    const float* xb   = x + (size_t)b * (NTOK * CCH);
    const int    widx = b & (NWIN - 1);       // window-mask index = b % NW

    // ---- padding bias + new mask ----
    if (tid < NTOK) {
        float m = mwin[b * NTOK + tid];
        sBias[tid] = (m == 0.f) ? -100.f : 0.f;
    }
    if (tid == 0) {
        float s = 0.f;
        for (int n = 0; n < NTOK; ++n) s += mwin[b * NTOK + n];
        sMn[0] = fminf(fmaxf(s, 0.f), 1.f);
    }

    // ---- row L2 norms (wave w handles rows 8w..8w+7) ----
    for (int j = 0; j < 8; ++j) {
        int r = wave * 8 + j;
        const float* row = xb + r * CCH;
        float s = 0.f;
        for (int i = lane; i < CCH; i += 32) { float v = row[i]; s += v * v; }
        for (int m = 16; m; m >>= 1) s += __shfl_xor(s, m, 32);
        if (lane == 0) sNrm[r] = 1.f / fmaxf(sqrtf(s), 1e-12f);
    }

    // ---- stage x into LDS as bf16 (vectorized) ----
    for (int i = tid * 4; i < NTOK * CCH; i += 256 * 4) {
        int r = i >> 9, c = i & (CCH - 1);
        const float4 v = *(const float4*)(xb + i);
        bf16_t* d = sX + r * LDX + c;
        d[0] = f2bf(v.x); d[1] = f2bf(v.y); d[2] = f2bf(v.z); d[3] = f2bf(v.w);
    }
    __syncthreads();

    if (tid < NTOK) mask_new[b * NTOK + tid] = sMn[0];

    // wave -> 16x16 tile for M=64,N=32 shaped GEMMs
    const int mb_w = wave >> 1;   // 0..3
    const int nb_w = wave & 1;    // 0..1

    for (int h = 0; h < HEADS; ++h) {
        // ---- (a) QKV projection for this head: [64x512] x [512x32] ----
        const int orow = h * HDIM + nb_w * 16 + l15;   // output channel
        const bf16_t* wq_p = wq + (size_t)orow * CCH;
        const bf16_t* wk_p = wk + (size_t)orow * CCH;
        const bf16_t* wv_p = wv + (size_t)orow * CCH;
        const float qb_s = qb[orow], kb_s = kb[orow], vb_s = vb[orow];

        v8f accq = v8f_zero(), acck = v8f_zero(), accv = v8f_zero();
        for (int kk = 0; kk < CCH / 32; ++kk) {
            const int k0 = kk * 32;
            if (kk + 1 < CCH / 32) {     // hint next weight slab into cache
                __builtin_prefetch(wq_p + k0 + 32, 0, 0);
                __builtin_prefetch(wk_p + k0 + 32, 0, 0);
                __builtin_prefetch(wv_p + k0 + 32, 0, 0);
            }
            v16bf a  = load_frag(sX, LDX, mb_w * 16, k0, lane);
            v16bf bq = load_frag(wq, CCH, h * HDIM + nb_w * 16, k0, lane);
            v16bf bk = load_frag(wk, CCH, h * HDIM + nb_w * 16, k0, lane);
            v16bf bv = load_frag(wv, CCH, h * HDIM + nb_w * 16, k0, lane);
            accq = wmma_bf16(a, bq, accq);
            acck = wmma_bf16(a, bk, acck);
            accv = wmma_bf16(a, bv, accv);
        }
        // scale rows by 1/||x|| (normalize folded post-GEMM), SCALE folded into q
        for (int r = 0; r < 8; ++r) {
            const int M    = mb_w * 16 + r + 8 * lhi;    // token
            const int ncol = nb_w * 16 + l15;            // head-dim index
            const float sc = sNrm[M];
            sQ[M * LDQ + ncol]  = f2bf((accq[r] + qb_s) * sc * SCALE_F);
            sK[M * LDQ + ncol]  = f2bf((acck[r] + kb_s) * sc);
            sVt[ncol * LDVT + M] = f2bf(accv[r] + vb_s); // store V transposed
        }
        __syncthreads();

        // ---- (b) scores: q @ k^T (K = 32), + window mask + padding bias ----
        const float* mrow = mask + (size_t)widx * NTOK * NTOK;
        for (int t = wave; t < 16; t += 8) {
            const int mb = t >> 2, nb = t & 3;
            v16bf qa = load_frag(sQ, LDQ, mb * 16, 0, lane);
            v16bf kbfr = load_frag(sK, LDQ, nb * 16, 0, lane); // B col n = k row n
            v8f s = wmma_bf16(qa, kbfr, v8f_zero());
            for (int r = 0; r < 8; ++r) {
                const int M = mb * 16 + r + 8 * lhi;
                const int Nn = nb * 16 + l15;
                sAttn[M * LDA + Nn] = s[r] + mrow[M * NTOK + Nn] + sBias[Nn];
            }
        }
        __syncthreads();

        // ---- (c) softmax over rows (wave32: lane owns cols lane, lane+32) ----
        for (int j = 0; j < 8; ++j) {
            const int r = wave * 8 + j;
            float* row = sAttn + r * LDA;
            float a0 = row[lane], a1 = row[lane + 32];
            float mx = fmaxf(a0, a1);
            for (int m = 16; m; m >>= 1) mx = fmaxf(mx, __shfl_xor(mx, m, 32));
            float e0 = __expf(a0 - mx), e1 = __expf(a1 - mx);
            float sm = e0 + e1;
            for (int m = 16; m; m >>= 1) sm += __shfl_xor(sm, m, 32);
            float inv = 1.f / sm;
            sP[r * LDP + lane]      = f2bf(e0 * inv);
            sP[r * LDP + lane + 32] = f2bf(e1 * inv);
        }
        __syncthreads();

        // ---- (d) out_h = P @ V  (K = 64, V^T gives contiguous B frags) ----
        v8f o = v8f_zero();
        for (int kk = 0; kk < 2; ++kk) {
            v16bf pa = load_frag(sP, LDP, mb_w * 16, kk * 32, lane);
            v16bf vf = load_frag(sVt, LDVT, nb_w * 16, kk * 32, lane);
            o = wmma_bf16(pa, vf, o);
        }
        for (int r = 0; r < 8; ++r) {
            const int M = mb_w * 16 + r + 8 * lhi;
            const int d = nb_w * 16 + l15;
            sO[M * LDX + h * HDIM + d] = f2bf(o[r]);
        }
        __syncthreads();   // protect sQ/sK/sVt/sP reuse next head
    }

    // ---- output projection: [64x512] x [512x512], fp32 out + bias ----
    for (int nb = wave; nb < CCH / 16; nb += 8) {
        const int ocol = nb * 16 + l15;
        const float pbias = pb[ocol];
        for (int mb = 0; mb < 4; ++mb) {
            v8f acc = v8f_zero();
            for (int kk = 0; kk < CCH / 32; ++kk) {
                v16bf a  = load_frag(sO, LDX, mb * 16, kk * 32, lane);
                v16bf bw = load_frag(wp, CCH, nb * 16, kk * 32, lane);
                acc = wmma_bf16(a, bw, acc);
            }
            for (int r = 0; r < 8; ++r) {
                const int M = mb * 16 + r + 8 * lhi;
                out[((size_t)b * NTOK + M) * CCH + ocol] = acc[r] + pbias;
            }
        }
    }
}

// ---------------- host launcher ----------------
extern "C" void kernel_launch(void* const* d_in, const int* in_sizes, int n_in,
                              void* d_out, int out_size, void* d_ws, size_t ws_size,
                              hipStream_t stream) {
    (void)in_sizes; (void)n_in; (void)out_size; (void)ws_size;
    const float* x    = (const float*)d_in[0];
    const float* mwin = (const float*)d_in[1];
    const float* mask = (const float*)d_in[2];
    const float* qw   = (const float*)d_in[3];
    const float* qb   = (const float*)d_in[4];
    const float* kw   = (const float*)d_in[5];
    const float* kbv  = (const float*)d_in[6];
    const float* vw   = (const float*)d_in[7];
    const float* vb   = (const float*)d_in[8];
    const float* pw   = (const float*)d_in[9];
    const float* pb   = (const float*)d_in[10];

    float* out      = (float*)d_out;
    float* mask_new = out + (size_t)B_TOT * NTOK * CCH;

    bf16_t* wsb = (bf16_t*)d_ws;   // 4 x 512x512 bf16 = 2 MB
    cvt_weights_kernel<<<(CCH * CCH + 255) / 256, 256, 0, stream>>>(qw, kw, vw, pw, wsb);

    const size_t shmem =
        (size_t)(2 * NTOK * LDX + 2 * NTOK * LDQ + HDIM * LDVT + NTOK * LDP) * sizeof(bf16_t) +
        (size_t)(NTOK * LDA + NTOK + NTOK + 4) * sizeof(float);

    win_attn_kernel<<<B_TOT, 256, shmem, stream>>>(
        x, mwin, mask,
        wsb,                 qb,
        wsb + CCH * CCH,     kbv,
        wsb + 2 * CCH * CCH, vb,
        wsb + 3 * CCH * CCH, pb,
        out, mask_new);
}
